// LSTMTorch_67886253080993
// MI455X (gfx1250) — compile-verified, matrix operand order checked by reading, not run on hardware
//
#include <hip/hip_runtime.h>

// LSTM forward on MI455X (gfx1250, wave32, WMMA).
// T=512, B=64, D=1024, H=1024. Strategy:
//   1) convert Wx/Wh fp32 -> bf16 (workspace)
//   2) xg[T*B,4H] = x @ Wx^T + bh  (parallel bf16 WMMA GEMM, fp32 accum)
//   3) 512 sequential step kernels: gates = xg_t + h @ Wh^T (bf16 WMMA,
//      acc preloaded with xg tile), activations, c/h update fused.
// Wh(bf16)=8MB stays in the 192MB L2 across all 512 steps.
// Step kernel double-buffers the h tile in LDS using async global->LDS
// copies (ASYNCcnt) so staging overlaps the WMMA chain.

#define T_STEPS 512
#define BATCH   64
#define DDIM    1024
#define HDIM    1024
#define GDIM    4096   // 4*H

typedef __bf16 bf16;
typedef bf16  v16bf __attribute__((ext_vector_type(16)));
typedef float v8f   __attribute__((ext_vector_type(8)));
typedef int   v4i   __attribute__((ext_vector_type(4)));

#define AS1 __attribute__((address_space(1)))
#define AS3 __attribute__((address_space(3)))

#if __has_builtin(__builtin_amdgcn_global_load_async_to_lds_b128)
#define USE_ASYNC_LDS 1
#else
#define USE_ASYNC_LDS 0
#endif

template <int N>
__device__ __forceinline__ void wait_async() {
#if USE_ASYNC_LDS
#if __has_builtin(__builtin_amdgcn_s_wait_asynccnt)
    __builtin_amdgcn_s_wait_asynccnt(N);
#else
    asm volatile("s_wait_asynccnt %0" ::"i"(N) : "memory");
#endif
#endif
}

__device__ __forceinline__ float sigmoid_f(float x) {
    return 1.0f / (1.0f + __expf(-x));
}

// stage one 64x32 bf16 h-tile into an LDS buffer (128 threads cooperate)
__device__ __forceinline__ void stage_h(const bf16* __restrict__ h_in, bf16* lbuf,
                                        int tid, int kk) {
    int b = tid >> 1, seg = tid & 1;
    const bf16* g = h_in + (size_t)b * HDIM + kk + seg * 16;
    bf16* l = lbuf + b * 32 + seg * 16;
#if USE_ASYNC_LDS
    __builtin_amdgcn_global_load_async_to_lds_b128((AS1 v4i*)g, (AS3 v4i*)l, 0, 0);
    __builtin_amdgcn_global_load_async_to_lds_b128((AS1 v4i*)(g + 8), (AS3 v4i*)(l + 8), 0, 0);
#else
    *(v16bf*)l = *(const v16bf*)g;
#endif
}

// ---------------------------------------------------------------- convert
__global__ __launch_bounds__(256) void cvt_f32_bf16(const float* __restrict__ src,
                                                    bf16* __restrict__ dst, int n) {
    int i = (blockIdx.x * blockDim.x + threadIdx.x) * 4;
    if (i + 3 < n) {
        float4 v = *(const float4*)(src + i);
        dst[i + 0] = (bf16)v.x;
        dst[i + 1] = (bf16)v.y;
        dst[i + 2] = (bf16)v.z;
        dst[i + 3] = (bf16)v.w;
    }
}

// ---------------------------------------------------------------- init h0/c0
__global__ __launch_bounds__(256) void init_state(float* __restrict__ c,
                                                  bf16* __restrict__ h0, int n) {
    int i = blockIdx.x * blockDim.x + threadIdx.x;
    if (i < n) {
        c[i]  = 0.0f;
        h0[i] = (bf16)0.0f;
    }
}

// ---------------------------------------------------------------- xg = x@Wx^T + bh
// Block tile: 128(M) x 64(N), 256 threads = 8 waves, each wave 64x16 (4 WMMA accs).
__global__ __launch_bounds__(256) void xproj_gemm(const float* __restrict__ x,     // [T*B, D]
                                                  const bf16* __restrict__ Wxb,    // [4H, D]
                                                  const float* __restrict__ bh,    // [4H]
                                                  float* __restrict__ xg) {        // [T*B, 4H]
    __shared__ __align__(32) bf16 lds_x[128 * 32];
    __shared__ __align__(32) bf16 lds_w[64 * 32];

    const int m0   = blockIdx.y * 128;
    const int n0   = blockIdx.x * 64;
    const int tid  = threadIdx.x;
    const int wave = tid >> 5;
    const int lane = tid & 31;
    const int wm   = wave >> 2;          // 0..1 : M half
    const int wn   = wave & 3;           // 0..3 : 16-col group
    const int lhalf = lane >> 4;         // 0..1
    const int l16   = lane & 15;

    v8f acc[4] = {};

    for (int kk = 0; kk < DDIM; kk += 32) {
        // stage x tile 128x32 (fp32 -> bf16 on the fly)
        {
            int row = tid >> 1, seg = tid & 1;
            const float* p = x + (size_t)(m0 + row) * DDIM + kk + seg * 16;
            bf16* q = &lds_x[row * 32 + seg * 16];
#pragma unroll
            for (int j = 0; j < 4; ++j) {
                float4 v = *(const float4*)(p + j * 4);
                q[j * 4 + 0] = (bf16)v.x;
                q[j * 4 + 1] = (bf16)v.y;
                q[j * 4 + 2] = (bf16)v.z;
                q[j * 4 + 3] = (bf16)v.w;
            }
        }
        // stage Wx tile 64x32 (already bf16: 16B copies)
        {
            int n = tid >> 2, seg = tid & 3;
            *(float4*)&lds_w[n * 32 + seg * 8] =
                *(const float4*)(Wxb + (size_t)(n0 + n) * DDIM + kk + seg * 8);
        }
        __syncthreads();

        v16bf bfrag = *(const v16bf*)&lds_w[(wn * 16 + l16) * 32 + lhalf * 16];
        const bf16* A = &lds_x[(wm * 64 + l16) * 32 + lhalf * 16];
        v16bf a0 = *(const v16bf*)(A + 0 * 16 * 32);
        v16bf a1 = *(const v16bf*)(A + 1 * 16 * 32);
        v16bf a2 = *(const v16bf*)(A + 2 * 16 * 32);
        v16bf a3 = *(const v16bf*)(A + 3 * 16 * 32);
        acc[0] = __builtin_amdgcn_wmma_f32_16x16x32_bf16(false, a0, false, bfrag, (short)0, acc[0], false, false);
        acc[1] = __builtin_amdgcn_wmma_f32_16x16x32_bf16(false, a1, false, bfrag, (short)0, acc[1], false, false);
        acc[2] = __builtin_amdgcn_wmma_f32_16x16x32_bf16(false, a2, false, bfrag, (short)0, acc[2], false, false);
        acc[3] = __builtin_amdgcn_wmma_f32_16x16x32_bf16(false, a3, false, bfrag, (short)0, acc[3], false, false);
        __syncthreads();
    }

    // epilogue: C layout => VGPR r holds M=r (lanes 0-15), M=r+8 (lanes 16-31); N=lane%16
    const int col = n0 + wn * 16 + l16;
    const float b = bh[col];
#pragma unroll
    for (int m = 0; m < 4; ++m) {
#pragma unroll
        for (int r = 0; r < 8; ++r) {
            int row = m0 + wm * 64 + m * 16 + r + lhalf * 8;
            xg[(size_t)row * GDIM + col] = acc[m][r] + b;
        }
    }
}

// ---------------------------------------------------------------- one LSTM step
// grid = H/16 blocks; each block owns h-columns [j*16, j*16+16) of ALL 4 gates.
// 4 waves = 4 gates; each wave: 64(batch) x 16(cols) tile = 4 WMMA accumulators.
// h tile double-buffered in LDS via async global->LDS; B frags pipelined from L2.
__global__ __launch_bounds__(128) void lstm_step(const float* __restrict__ xg_t,  // [B,4H]
                                                 const bf16* __restrict__ Whb,    // [4H,H]
                                                 const bf16* __restrict__ h_in,   // [B,H]
                                                 bf16* __restrict__ h_out,        // [B,H]
                                                 float* __restrict__ c,           // [B,H]
                                                 float* __restrict__ out_t) {     // [B,H]
    __shared__ __align__(32) bf16 lds_h[2][64 * 32];
    __shared__ float lds_g[4][64][16];

    const int j    = blockIdx.x;
    const int tid  = threadIdx.x;
    const int g    = tid >> 5;           // gate index = wave id (0:i 1:f 2:g 3:o)
    const int lane = tid & 31;
    const int lhalf = lane >> 4;
    const int l16   = lane & 15;
    const int coln  = j * 16 + l16;      // column within H for this lane
    const int ng    = g * HDIM + coln;   // row of Wh / column of gates

    // preload accumulators with xg tile (C-matrix VGPR layout)
    v8f acc[4];
#pragma unroll
    for (int m = 0; m < 4; ++m) {
#pragma unroll
        for (int r = 0; r < 8; ++r) {
            int b = m * 16 + r + lhalf * 8;
            acc[m][r] = xg_t[(size_t)b * GDIM + ng];
        }
    }

    const bf16* wrow = Whb + (size_t)ng * HDIM;  // per-lane Wh row (resident in L2)

    constexpr int KITERS = HDIM / 32;
    stage_h(h_in, lds_h[0], tid, 0);                       // prologue stage
    v16bf bfrag = *(const v16bf*)(wrow + lhalf * 16);      // prologue B frag

    for (int i = 0; i < KITERS; ++i) {
        const int kk = i * 32;
        if (i + 1 < KITERS) {
            stage_h(h_in, lds_h[(i + 1) & 1], tid, kk + 32);   // overlap next stage
            wait_async<2>();                                    // current stage done
        } else {
            wait_async<0>();
        }
        __syncthreads();

        v16bf bnext = bfrag;
        if (i + 1 < KITERS)
            bnext = *(const v16bf*)(wrow + kk + 32 + lhalf * 16);

        const bf16* A = &lds_h[i & 1][l16 * 32 + lhalf * 16];
        v16bf a0 = *(const v16bf*)(A + 0 * 16 * 32);
        v16bf a1 = *(const v16bf*)(A + 1 * 16 * 32);
        v16bf a2 = *(const v16bf*)(A + 2 * 16 * 32);
        v16bf a3 = *(const v16bf*)(A + 3 * 16 * 32);
        acc[0] = __builtin_amdgcn_wmma_f32_16x16x32_bf16(false, a0, false, bfrag, (short)0, acc[0], false, false);
        acc[1] = __builtin_amdgcn_wmma_f32_16x16x32_bf16(false, a1, false, bfrag, (short)0, acc[1], false, false);
        acc[2] = __builtin_amdgcn_wmma_f32_16x16x32_bf16(false, a2, false, bfrag, (short)0, acc[2], false, false);
        acc[3] = __builtin_amdgcn_wmma_f32_16x16x32_bf16(false, a3, false, bfrag, (short)0, acc[3], false, false);

        __syncthreads();   // all reads of this buffer done before it is re-staged
        bfrag = bnext;
    }

    // activation per gate, exchange through LDS
#pragma unroll
    for (int m = 0; m < 4; ++m) {
#pragma unroll
        for (int r = 0; r < 8; ++r) {
            float v = acc[m][r];
            v = (g == 2) ? tanhf(v) : sigmoid_f(v);
            int b = m * 16 + r + lhalf * 8;
            lds_g[g][b][l16] = v;
        }
    }
    __syncthreads();

    // elementwise state update: 64x16 = 1024 elements / 128 threads = 8 each
#pragma unroll
    for (int e = 0; e < 8; ++e) {
        int q = tid + 128 * e;
        int b = q >> 4, n = q & 15;
        int col = j * 16 + n;
        float iv = lds_g[0][b][n];
        float fv = lds_g[1][b][n];
        float gv = lds_g[2][b][n];
        float ov = lds_g[3][b][n];
        size_t idx = (size_t)b * HDIM + col;
        float cnew = fv * c[idx] + iv * gv;
        float hv   = ov * tanhf(cnew);
        c[idx]     = cnew;
        out_t[idx] = hv;
        h_out[idx] = (bf16)hv;
    }
}

// ---------------------------------------------------------------- launch
extern "C" void kernel_launch(void* const* d_in, const int* in_sizes, int n_in,
                              void* d_out, int out_size, void* d_ws, size_t ws_size,
                              hipStream_t stream) {
    const float* x  = (const float*)d_in[0];   // [T,B,D]
    const float* Wx = (const float*)d_in[1];   // [4H,D]
    const float* Wh = (const float*)d_in[2];   // [4H,H]
    const float* bh = (const float*)d_in[3];   // [4H]
    float* out = (float*)d_out;                // outputs[T,B,H] | h[B,H] | c[B,H]

    char* ws = (char*)d_ws;
    bf16* Wxb = (bf16*)ws;  ws += (size_t)GDIM * DDIM * sizeof(bf16);
    bf16* Whb = (bf16*)ws;  ws += (size_t)GDIM * HDIM * sizeof(bf16);
    float* xg = (float*)ws; ws += (size_t)T_STEPS * BATCH * GDIM * sizeof(float);
    float* c  = (float*)ws; ws += (size_t)BATCH * HDIM * sizeof(float);
    bf16* h0  = (bf16*)ws;  ws += (size_t)BATCH * HDIM * sizeof(bf16);
    bf16* h1  = (bf16*)ws;

    {
        int n = GDIM * DDIM;
        cvt_f32_bf16<<<n / 1024, 256, 0, stream>>>(Wx, Wxb, n);
        n = GDIM * HDIM;
        cvt_f32_bf16<<<n / 1024, 256, 0, stream>>>(Wh, Whb, n);
    }
    init_state<<<(BATCH * HDIM) / 256, 256, 0, stream>>>(c, h0, BATCH * HDIM);

    xproj_gemm<<<dim3(GDIM / 64, (T_STEPS * BATCH) / 128), 256, 0, stream>>>(x, Wxb, bh, xg);

    for (int t = 0; t < T_STEPS; ++t) {
        bf16* hin  = (t & 1) ? h1 : h0;
        bf16* hout = (t & 1) ? h0 : h1;
        lstm_step<<<HDIM / 16, 128, 0, stream>>>(
            xg + (size_t)t * BATCH * GDIM, Whb, hin, hout, c,
            out + (size_t)t * BATCH * HDIM);
    }

    // final h = outputs[T-1]; final c from workspace
    (void)hipMemcpyAsync(out + (size_t)T_STEPS * BATCH * HDIM,
                         out + (size_t)(T_STEPS - 1) * BATCH * HDIM,
                         (size_t)BATCH * HDIM * sizeof(float),
                         hipMemcpyDeviceToDevice, stream);
    (void)hipMemcpyAsync(out + (size_t)T_STEPS * BATCH * HDIM + (size_t)BATCH * HDIM,
                         c,
                         (size_t)BATCH * HDIM * sizeof(float),
                         hipMemcpyDeviceToDevice, stream);
}
